// SwinTransformerBlock_47090021433554
// MI455X (gfx1250) — matrix operand order repaired and checked by reading, not compiled, not run on hardware
//
#include <hip/hip_runtime.h>
#include <hip/hip_bf16.h>
#include <math.h>

// ---------------------------------------------------------------------------
// Swin block on gfx1250: bf16 WMMA GEMMs (async-to-LDS double-buffered,
// LDS-restaged coalesced bf16 epilogues) + WMMA windowed attention.
// ---------------------------------------------------------------------------

typedef __bf16 bf16_t;
typedef __attribute__((ext_vector_type(16))) __bf16 v16bf;
typedef __attribute__((ext_vector_type(8)))  float  v8f;
typedef __attribute__((ext_vector_type(4)))  int    v4i_t;

#define AS1 __attribute__((address_space(1)))
#define AS3 __attribute__((address_space(3)))

#define BATCH   16
#define HR      64
#define WR      64
#define CH      512
#define NHEAD   16
#define WSZ     8
#define NTOK    64          // tokens per window
#define TABSZ   15          // 2*WSZ-1
#define DHEAD   32
#define NWIN    1024        // BATCH * (HR/WSZ) * (WR/WSZ)
#define MROWS   65536       // BATCH*HR*WR

// ---- window index permutations (folded into epilogues) --------------------
__device__ __forceinline__ int win2tok(int r) {
    int w = r >> 6, n = r & 63;
    int b = w >> 6, wy = (w >> 3) & 7, wx = w & 7;
    int iy = n >> 3, ix = n & 7;
    return (b << 12) | (((wy << 3) | iy) << 6) | ((wx << 3) | ix);
}
__device__ __forceinline__ int tok2win(int t) {
    int b = t >> 12, y = (t >> 6) & 63, x = t & 63;
    int wy = y >> 3, iy = y & 7, wx = x >> 3, ix = x & 7;
    return ((((b << 6) | (wy << 3) | wx)) << 6) | ((iy << 3) | ix);
}

// ---- fragment loaders (layouts per CDNA5 ISA) -------------------
union FragU { v16bf v; uint4 u[2]; };

// A 16x32 bf16: lane holds M=lane&15; elems 0-7 = K(off..off+7), 8-15 = K(+16)
__device__ __forceinline__ v16bf load_frag_a(const bf16_t* p) {
    FragU f;
    f.u[0] = *(const uint4*)p;
    f.u[1] = *(const uint4*)(p + 16);
    return f.v;
}
// B 32x16 bf16: lane holds N=lane&15; elems 0-7 = K(off..off+7), 8-15 = K(+16)
__device__ __forceinline__ v16bf load_frag_b(const bf16_t* p) {
    FragU f;
    f.u[0] = *(const uint4*)p;
    f.u[1] = *(const uint4*)(p + 16);
    return f.v;
}

// ---- swizzled LDS tile helpers (128 rows x 64 bf16, 8 chunks of 16B/row) ---
__device__ __forceinline__ int row_swz(int row) { return (row ^ (row >> 3)) & 7; }

// A fragment from swizzled tile: chunks (kc*4+half) and (+2)
__device__ __forceinline__ v16bf frag_a_sw(const bf16_t* s, int row, int kc, int half) {
    const int sw = row_swz(row);
    const int c0 = (kc * 4 + half) ^ sw;
    const int c1 = (kc * 4 + half + 2) ^ sw;
    FragU f;
    f.u[0] = *(const uint4*)(s + row * 64 + (c0 << 3));
    f.u[1] = *(const uint4*)(s + row * 64 + (c1 << 3));
    return f.v;
}
// B fragment: chunks (kc*4+half) and (+2)
__device__ __forceinline__ v16bf frag_b_sw(const bf16_t* s, int row, int kc, int half) {
    const int sw = row_swz(row);
    const int c0 = (kc * 4 + half) ^ sw;
    const int c1 = (kc * 4 + half + 2) ^ sw;
    FragU f;
    f.u[0] = *(const uint4*)(s + row * 64 + (c0 << 3));
    f.u[1] = *(const uint4*)(s + row * 64 + (c1 << 3));
    return f.v;
}

// async global->LDS staging of one 128x64 bf16 tile (4 x b128 per thread)
__device__ __forceinline__ void stage_tile_async(const bf16_t* __restrict__ G, int ldg,
                                                 int base_row, int k0,
                                                 bf16_t* s, int tid) {
#pragma unroll
    for (int i = 0; i < 4; ++i) {
        const int c = tid + i * 256;            // 0..1023 chunk id
        const int row = c >> 3, ch = c & 7;
        const int sw = ch ^ row_swz(row);
        __builtin_amdgcn_global_load_async_to_lds_b128(
            (AS1 v4i_t*)(G + (size_t)(base_row + row) * ldg + k0 + (ch << 3)),
            (AS3 v4i_t*)(s + row * 64 + (sw << 3)),
            0, 0);
    }
}

// ---------------------------------------------------------------------------
// Generic bf16 WMMA GEMM:  out = epilogue(A[MxK] @ Bw[NxK]^T + bias)
// Block tile 128x128, BK=64, 256 threads (8 waves), wave tile 64x32.
// Double-buffered LDS filled by GLOBAL_LOAD_ASYNC_TO_LDS_B128 (ASYNCcnt).
// bf16 outputs (EPI 0/2) are restaged through LDS for coalesced b128 stores.
// EPI: 0 = QKV (scale q cols, bf16 out)
//      1 = proj (window-reverse + residual x, f32 out)
//      2 = fc1  (GELU, bf16 out)
//      3 = fc2  (+residual in-place, f32 out)
// ---------------------------------------------------------------------------
template<int EPI>
__global__ __launch_bounds__(256)
void gemm_wmma(const bf16_t* __restrict__ A, const bf16_t* __restrict__ Bw,
               const float* __restrict__ bias, const float* __restrict__ resid,
               bf16_t* __restrict__ out_bf, float* __restrict__ out_f,
               int M, int N, int K, float qscale)
{
    __shared__ bf16_t As[2][128 * 64];
    __shared__ bf16_t Bs[2][128 * 64];        // 64 KB total

    const int tid  = threadIdx.x;
    const int bm   = blockIdx.y * 128;
    const int bn   = blockIdx.x * 128;
    const int wv   = tid >> 5;
    const int lane = tid & 31;
    const int wm   = wv >> 2;                 // 0..1
    const int wn   = wv & 3;                  // 0..3
    const int half = lane >> 4;
    const int l15  = lane & 15;

    v8f acc[4][2] = {};

    // prologue: stage first tiles
    stage_tile_async(A,  K, bm, 0, As[0], tid);
    stage_tile_async(Bw, K, bn, 0, Bs[0], tid);
    __builtin_amdgcn_s_wait_asynccnt(0);
    __syncthreads();

    int buf = 0;
    for (int k0 = 0; k0 < K; k0 += 64) {
        if (k0 + 64 < K) {                    // prefetch next tiles into other buffer
            stage_tile_async(A,  K, bm, k0 + 64, As[buf ^ 1], tid);
            stage_tile_async(Bw, K, bn, k0 + 64, Bs[buf ^ 1], tid);
        }
#pragma unroll
        for (int kc = 0; kc < 2; ++kc) {
            v16bf af[4], bfg[2];
#pragma unroll
            for (int tm = 0; tm < 4; ++tm)
                af[tm] = frag_a_sw(As[buf], wm * 64 + tm * 16 + l15, kc, half);
#pragma unroll
            for (int tn = 0; tn < 2; ++tn)
                bfg[tn] = frag_b_sw(Bs[buf], wn * 32 + tn * 16 + l15, kc, half);
#pragma unroll
            for (int tm = 0; tm < 4; ++tm)
#pragma unroll
                for (int tn = 0; tn < 2; ++tn)
                    acc[tm][tn] = __builtin_amdgcn_wmma_f32_16x16x32_bf16(
                        false, af[tm], false, bfg[tn], (short)0, acc[tm][tn],
                        false, false);
        }
        __builtin_amdgcn_s_wait_asynccnt(0);  // my prefetch landed
        __syncthreads();                      // everyone done reading buf
        buf ^= 1;
    }

    // ---- epilogue (C layout: N across lanes, M across vector elems) -------
    if constexpr (EPI == 0 || EPI == 2) {
        // bf16 outputs: restage C tile through (now dead) staging LDS, then
        // stream out coalesced 128B row segments.
        bf16_t* Cs = (bf16_t*)As;             // 128 x 128 bf16 = 32 KB
#pragma unroll
        for (int tm = 0; tm < 4; ++tm)
#pragma unroll
            for (int tn = 0; tn < 2; ++tn) {
                const int nl = wn * 32 + tn * 16 + l15;
                const float bv = bias[bn + nl];
#pragma unroll
                for (int j = 0; j < 8; ++j) {
                    const int ml = wm * 64 + tm * 16 + half * 8 + j;
                    float v = acc[tm][tn][j] + bv;
                    if constexpr (EPI == 0) {
                        if (bn + nl < CH) v *= qscale;            // scale q part
                    } else {
                        v = 0.5f * v * (1.0f + erff(v * 0.70710678118654752f));
                    }
                    Cs[ml * 128 + nl] = (bf16_t)v;
                }
            }
        __syncthreads();
        const int row  = tid >> 1;
        const int colh = (tid & 1) << 6;      // 0 or 64
        const uint4* src = (const uint4*)(Cs + row * 128 + colh);
        uint4* dst = (uint4*)(out_bf + (size_t)(bm + row) * N + bn + colh);
#pragma unroll
        for (int q = 0; q < 8; ++q) dst[q] = src[q];
    } else {
        // f32 outputs: direct stores are 64B-contiguous per half-wave.
#pragma unroll
        for (int tm = 0; tm < 4; ++tm)
#pragma unroll
            for (int tn = 0; tn < 2; ++tn) {
                const int n  = bn + wn * 32 + tn * 16 + l15;
                const float bv = bias[n];
#pragma unroll
                for (int j = 0; j < 8; ++j) {
                    const int m = bm + wm * 64 + tm * 16 + half * 8 + j;
                    float v = acc[tm][tn][j] + bv;
                    if constexpr (EPI == 1) {         // proj + residual, win-reverse
                        const int t = win2tok(m);
                        v += resid[(size_t)t * CH + n];
                        out_f[(size_t)t * CH + n] = v;
                    } else {                          // fc2 + residual in-place
                        const float r = resid[(size_t)m * CH + n];
                        out_f[(size_t)m * CH + n] = v + r;
                    }
                }
            }
    }
}

// ---------------------------------------------------------------------------
// Windowed attention: one wave per (window, head). 4 waves / block.
// S = (q*scale) K^T  (16 WMMA, K=DH=32) ; +bias ; softmax ; O = P V (16 WMMA)
// ---------------------------------------------------------------------------
__global__ __launch_bounds__(128)
void attn_wmma(const bf16_t* __restrict__ qkv, const float* __restrict__ biasT,
               bf16_t* __restrict__ out)
{
    const int LDP = 72;                               // padded row (bf16)
    __shared__ bf16_t lds[4 * (32 * 72 + 64 * 72)];   // per-wave Vt + P

    const int wv   = threadIdx.x >> 5;
    const int lane = threadIdx.x & 31;
    const int wh   = blockIdx.x * 4 + wv;
    const int w    = wh >> 4;
    const int h    = wh & 15;
    const int half = lane >> 4;
    const int l15  = lane & 15;

    bf16_t* Vt = &lds[wv * (32 * 72 + 64 * 72)];
    bf16_t* P  = Vt + 32 * 72;

    const size_t rs = 3 * CH;                         // 1536 row stride
    const bf16_t* qb = qkv + (size_t)w * 64 * rs + h * DHEAD;
    const bf16_t* kb = qb + CH;
    const bf16_t* vb = qb + 2 * CH;

    // ---- S = Q K^T : A frags from q rows, B frags from k rows (contig in d)
    v16bf qf[4], kf[4];
#pragma unroll
    for (int tm = 0; tm < 4; ++tm)
        qf[tm] = load_frag_a(qb + (size_t)(tm * 16 + l15) * rs + half * 8);
#pragma unroll
    for (int tn = 0; tn < 4; ++tn)
        kf[tn] = load_frag_b(kb + (size_t)(tn * 16 + l15) * rs + half * 8);

    v8f s[4][4] = {};
#pragma unroll
    for (int tm = 0; tm < 4; ++tm)
#pragma unroll
        for (int tn = 0; tn < 4; ++tn)
            s[tm][tn] = __builtin_amdgcn_wmma_f32_16x16x32_bf16(
                false, qf[tm], false, kf[tn], (short)0, s[tm][tn], false, false);

    // ---- stage V transposed into LDS: Vt[d][kv]
#pragma unroll
    for (int rr = 0; rr < 2; ++rr) {
        const int r = lane * 2 + rr;
        const bf16_t* vp = vb + (size_t)r * rs;
#pragma unroll
        for (int d = 0; d < 32; ++d) Vt[d * LDP + r] = vp[d];
    }
    __syncthreads();

    // ---- bias + row softmax (row = one 16-lane half, 4 tn tiles)
    const float* bh = biasT + (size_t)h * 64 * 64;
#pragma unroll
    for (int tm = 0; tm < 4; ++tm) {
#pragma unroll
        for (int j = 0; j < 8; ++j) {
            const int m = tm * 16 + half * 8 + j;
            float vv[4];
            float mx = -3.0e38f;
#pragma unroll
            for (int tn = 0; tn < 4; ++tn) {
                const int n = tn * 16 + l15;
                float v = s[tm][tn][j] + bh[m * 64 + n];
                vv[tn] = v;
                mx = fmaxf(mx, v);
            }
            for (int d = 1; d < 16; d <<= 1)
                mx = fmaxf(mx, __shfl_xor(mx, d, 32));
            float sum = 0.f;
#pragma unroll
            for (int tn = 0; tn < 4; ++tn) { vv[tn] = __expf(vv[tn] - mx); sum += vv[tn]; }
            for (int d = 1; d < 16; d <<= 1)
                sum += __shfl_xor(sum, d, 32);
            const float inv = 1.0f / sum;
#pragma unroll
            for (int tn = 0; tn < 4; ++tn)
                P[m * LDP + tn * 16 + l15] = (bf16_t)(vv[tn] * inv);
        }
    }
    __syncthreads();

    // ---- O = P V : K = 64 -> 2 chunks of 32
    v8f o[4][2] = {};
#pragma unroll
    for (int kc = 0; kc < 2; ++kc) {
        v16bf pf[4], vf[2];
#pragma unroll
        for (int tm = 0; tm < 4; ++tm)
            pf[tm] = load_frag_a(P + (tm * 16 + l15) * LDP + kc * 32 + half * 8);
#pragma unroll
        for (int tn = 0; tn < 2; ++tn)
            vf[tn] = load_frag_b(Vt + (tn * 16 + l15) * LDP + kc * 32 + half * 8);
#pragma unroll
        for (int tm = 0; tm < 4; ++tm)
#pragma unroll
            for (int tn = 0; tn < 2; ++tn)
                o[tm][tn] = __builtin_amdgcn_wmma_f32_16x16x32_bf16(
                    false, pf[tm], false, vf[tn], (short)0, o[tm][tn], false, false);
    }

    // ---- store O (winrow-major, bf16 feeds proj GEMM)
#pragma unroll
    for (int tm = 0; tm < 4; ++tm)
#pragma unroll
        for (int tn = 0; tn < 2; ++tn)
#pragma unroll
            for (int j = 0; j < 8; ++j) {
                const int m = tm * 16 + half * 8 + j;
                const int d = tn * 16 + l15;
                out[(size_t)(w * 64 + m) * CH + h * DHEAD + d] = (bf16_t)o[tm][tn][j];
            }
}

// ---------------------------------------------------------------------------
// LayerNorm -> bf16 (optionally writing rows in window-partitioned order)
// ---------------------------------------------------------------------------
template<bool WIN>
__global__ __launch_bounds__(256)
void ln_kernel(const float* __restrict__ x, const float* __restrict__ g,
               const float* __restrict__ b, bf16_t* __restrict__ out)
{
    __shared__ float red[256];
    const int row = blockIdx.x;
    const int t   = threadIdx.x;
    const float v0 = x[(size_t)row * CH + t];
    const float v1 = x[(size_t)row * CH + t + 256];

    red[t] = v0 + v1;
    __syncthreads();
    for (int d = 128; d > 0; d >>= 1) { if (t < d) red[t] += red[t + d]; __syncthreads(); }
    const float mu = red[0] * (1.0f / 512.0f);
    __syncthreads();

    const float d0 = v0 - mu, d1 = v1 - mu;
    red[t] = d0 * d0 + d1 * d1;
    __syncthreads();
    for (int d = 128; d > 0; d >>= 1) { if (t < d) red[t] += red[t + d]; __syncthreads(); }
    const float rstd = rsqrtf(red[0] * (1.0f / 512.0f) + 1e-5f);

    const int orow = WIN ? tok2win(row) : row;
    out[(size_t)orow * CH + t]       = (bf16_t)(d0 * rstd * g[t]       + b[t]);
    out[(size_t)orow * CH + t + 256] = (bf16_t)(d1 * rstd * g[t + 256] + b[t + 256]);
}

// ---------------------------------------------------------------------------
// Precompute attention bias table biasT[h][i][j]  (16*64*64 f32, L2-resident)
// ---------------------------------------------------------------------------
__global__ __launch_bounds__(256)
void bias_table(const float* __restrict__ a_p, const float* __restrict__ b_p,
                const float* __restrict__ a_r, const float* __restrict__ b_r,
                const int* __restrict__ radius_idx, const int* __restrict__ azimuth_idx,
                const float* __restrict__ theta_max, float* __restrict__ biasT)
{
    const int idx = blockIdx.x * 256 + threadIdx.x;
    if (idx >= NHEAD * 64 * 64) return;
    const int h = idx >> 12, ij = idx & 4095;
    const int ai = azimuth_idx[ij], ri = radius_idx[ij];
    const float rel_a = (ai < WSZ) ? (float)ai : (float)(ai - TABSZ);
    const float rel_r = (ri < WSZ) ? (float)ri : (float)(ri - TABSZ);
    const float ph_a = rel_a * (6.283185307179586f / (float)WR);
    const float ph_r = rel_r * theta_max[0] * (1.0f / (float)HR);
    const float v = a_p[ai * NHEAD + h] * cosf(ph_a) + b_p[ai * NHEAD + h] * sinf(ph_a)
                  + a_r[ri * NHEAD + h] * cosf(ph_r) + b_r[ri * NHEAD + h] * sinf(ph_r);
    biasT[idx] = v;
}

__global__ __launch_bounds__(256)
void f2bf(const float* __restrict__ in, bf16_t* __restrict__ out, int n)
{
    const int i = blockIdx.x * 256 + threadIdx.x;
    if (i < n) out[i] = (bf16_t)in[i];
}

// ---------------------------------------------------------------------------
extern "C" void kernel_launch(void* const* d_in, const int* in_sizes, int n_in,
                              void* d_out, int out_size, void* d_ws, size_t ws_size,
                              hipStream_t stream)
{
    const float* x        = (const float*)d_in[0];
    const float* theta    = (const float*)d_in[1];
    const float* norm1_g  = (const float*)d_in[2];
    const float* norm1_b  = (const float*)d_in[3];
    const float* qkv_w    = (const float*)d_in[4];
    const float* qkv_b    = (const float*)d_in[5];
    const float* a_p      = (const float*)d_in[6];
    const float* b_p      = (const float*)d_in[7];
    const float* a_r      = (const float*)d_in[8];
    const float* b_r      = (const float*)d_in[9];
    const float* proj_w   = (const float*)d_in[10];
    const float* proj_b   = (const float*)d_in[11];
    const float* norm2_g  = (const float*)d_in[12];
    const float* norm2_b  = (const float*)d_in[13];
    const float* fc1_w    = (const float*)d_in[14];
    const float* fc1_b    = (const float*)d_in[15];
    const float* fc2_w    = (const float*)d_in[16];
    const float* fc2_b    = (const float*)d_in[17];
    const int*   rad_idx  = (const int*)d_in[18];
    const int*   azi_idx  = (const int*)d_in[19];

    char* ws = (char*)d_ws;
    bf16_t* h1   = (bf16_t*)ws;                              //  64 MB (h1, then attn_out)
    bf16_t* qkvB = (bf16_t*)(ws + ((size_t)64  << 20));      // 192 MB (qkv, then h2)
    bf16_t* fc1o = (bf16_t*)(ws + ((size_t)256 << 20));      // 256 MB
    char*   aux  = ws + ((size_t)512 << 20);
    bf16_t* qkvw_bf = (bf16_t*)aux;
    bf16_t* projw_bf = qkvw_bf + 1536 * 512;
    bf16_t* fc1w_bf  = projw_bf + 512 * 512;
    bf16_t* fc2w_bf  = fc1w_bf  + 2048 * 512;
    float*  biasT    = (float*)(fc2w_bf + 512 * 2048);

    bf16_t* attn_o = h1;       // reuse: h1 dead after GEMM1
    bf16_t* h2     = qkvB;     // reuse: qkv dead after attention
    float*  x2     = (float*)d_out;

    const float qscale = 0.17677669529663687f;   // DH^-0.5

    // weight conversions + bias table
    f2bf<<<(1536 * 512 + 255) / 256, 256, 0, stream>>>(qkv_w,  qkvw_bf, 1536 * 512);
    f2bf<<<(512  * 512 + 255) / 256, 256, 0, stream>>>(proj_w, projw_bf, 512 * 512);
    f2bf<<<(2048 * 512 + 255) / 256, 256, 0, stream>>>(fc1_w,  fc1w_bf, 2048 * 512);
    f2bf<<<(512 * 2048 + 255) / 256, 256, 0, stream>>>(fc2_w,  fc2w_bf, 512 * 2048);
    bias_table<<<(NHEAD * 64 * 64) / 256, 256, 0, stream>>>(a_p, b_p, a_r, b_r,
                                                            rad_idx, azi_idx, theta, biasT);

    // LN1 (token -> window order, bf16)
    ln_kernel<true><<<MROWS, 256, 0, stream>>>(x, norm1_g, norm1_b, h1);

    // QKV GEMM: 65536x1536x512
    gemm_wmma<0><<<dim3(1536 / 128, MROWS / 128), 256, 0, stream>>>(
        h1, qkvw_bf, qkv_b, nullptr, qkvB, nullptr, MROWS, 1536, 512, qscale);

    // windowed attention
    attn_wmma<<<(NWIN * NHEAD) / 4, 128, 0, stream>>>(qkvB, biasT, attn_o);

    // proj GEMM + window-reverse + residual -> x2 (d_out)
    gemm_wmma<1><<<dim3(512 / 128, MROWS / 128), 256, 0, stream>>>(
        attn_o, projw_bf, proj_b, x, nullptr, x2, MROWS, 512, 512, 1.0f);

    // LN2 (token order)
    ln_kernel<false><<<MROWS, 256, 0, stream>>>(x2, norm2_g, norm2_b, h2);

    // fc1 GEMM + GELU: 65536x2048x512
    gemm_wmma<2><<<dim3(2048 / 128, MROWS / 128), 256, 0, stream>>>(
        h2, fc1w_bf, fc1_b, nullptr, fc1o, nullptr, MROWS, 2048, 512, 1.0f);

    // fc2 GEMM + residual (in-place on d_out): 65536x512x2048
    gemm_wmma<3><<<dim3(512 / 128, MROWS / 128), 256, 0, stream>>>(
        fc1o, fc2w_bf, fc2_b, x2, nullptr, x2, MROWS, 512, 2048, 1.0f);
}
